// SimpleRNN_68899865362870
// MI455X (gfx1250) — matrix operand order repaired and checked by reading, not compiled, not run on hardware
//
#include <hip/hip_runtime.h>
#include <hip/hip_bf16.h>
#include <math.h>

// ---------------------------------------------------------------------------
// Two-layer Elman RNN, persistent-kernel formulation for MI455X (gfx1250).
//   h1 = tanh(emb[x_t] @ Wi1 + b1 + h1 @ Wh1)
//   h2 = tanh(h1       @ Wi2 + b2 + h2 @ Wh2)
//   out = sigmoid(h2_last @ Wd + bd)
// bf16 WMMA (v_wmma_f32_16x16x32_bf16) with fp32 accumulate; weights resident
// in LDS in WMMA B-fragment layout; device-wide barrier between timesteps.
// ---------------------------------------------------------------------------

typedef __attribute__((ext_vector_type(16))) __bf16 v16bf;
typedef __attribute__((ext_vector_type(8)))  float  v8f;

#define EMBD   512
#define HID    1024
#define BATCH  64
#define SEQ    512
#define NWG    64          // one workgroup per 16-column slice of HID
#define WAVES  4           // one wave32 per 16-row slice of BATCH
#define TPB    (WAVES * 32)

// d_ws layout (bytes)
#define WS_CNT   0                          // barrier counter (u32)
#define WS_H1B0  1024                       // h1 ping  (64*1024 bf16 = 128 KB)
#define WS_H1B1  (WS_H1B0 + 131072)         // h1 pong
#define WS_H2B0  (WS_H1B1 + 131072)         // h2 ping
#define WS_H2B1  (WS_H2B0 + 131072)         // h2 pong

// LDS (halves): per K-tile fragment = 32 lanes * 16 halves = 512 halves
#define LDS_WI1  0                          // 16 k-tiles (K=512)
#define LDS_WH1  (LDS_WI1 + 16 * 512)       // 32 k-tiles (K=1024)
#define LDS_WI2  (LDS_WH1 + 32 * 512)
#define LDS_WH2  (LDS_WI2 + 32 * 512)
#define LDS_HALVES (LDS_WH2 + 32 * 512)     // 57344 halves = 114688 bytes

// ---- device-wide barrier (monotonic counter; graph-replay safe via init) ---
__device__ __forceinline__ void grid_sync(unsigned* cnt, unsigned target) {
  __threadfence();                 // release: flush this wave's stores (global_wb)
  __syncthreads();
  if (threadIdx.x == 0) {
    __hip_atomic_fetch_add(cnt, 1u, __ATOMIC_ACQ_REL, __HIP_MEMORY_SCOPE_AGENT);
    while (__hip_atomic_load(cnt, __ATOMIC_ACQUIRE, __HIP_MEMORY_SCOPE_AGENT) < target)
      __builtin_amdgcn_s_sleep(1);
  }
  __syncthreads();
  __threadfence();                 // acquire: invalidate stale L0 lines (global_inv)
}

// ---- A-fragment (16x32, M=lane&15) from row-major bf16 state in global ----
__device__ __forceinline__ v16bf load_a_bf16(const __bf16* h, int row, int kt, int lane) {
  const int g0 = kt * 32 + ((lane < 16) ? 0 : 8);    // K: e<8 group
  const int g1 = kt * 32 + ((lane < 16) ? 16 : 24);  // K: e>=8 group
  union { uint4 q[2]; v16bf v; } u;
  u.q[0] = *(const uint4*)(h + (size_t)row * HID + g0);
  u.q[1] = *(const uint4*)(h + (size_t)row * HID + g1);
  return u.v;
}

// ---- A-fragment from a row-major fp32 embedding row (convert to bf16) -----
__device__ __forceinline__ v16bf load_a_f32(const float* rowp, int kt, int lane) {
  const int g0 = kt * 32 + ((lane < 16) ? 0 : 8);
  const int g1 = kt * 32 + ((lane < 16) ? 16 : 24);
  float4 f0 = *(const float4*)(rowp + g0);
  float4 f1 = *(const float4*)(rowp + g0 + 4);
  float4 f2 = *(const float4*)(rowp + g1);
  float4 f3 = *(const float4*)(rowp + g1 + 4);
  v16bf v;
  v[0]  = (__bf16)f0.x; v[1]  = (__bf16)f0.y; v[2]  = (__bf16)f0.z; v[3]  = (__bf16)f0.w;
  v[4]  = (__bf16)f1.x; v[5]  = (__bf16)f1.y; v[6]  = (__bf16)f1.z; v[7]  = (__bf16)f1.w;
  v[8]  = (__bf16)f2.x; v[9]  = (__bf16)f2.y; v[10] = (__bf16)f2.z; v[11] = (__bf16)f2.w;
  v[12] = (__bf16)f3.x; v[13] = (__bf16)f3.y; v[14] = (__bf16)f3.z; v[15] = (__bf16)f3.w;
  return v;
}

// ---- B-fragment (32x16) read straight from fragment-packed LDS ------------
__device__ __forceinline__ v16bf load_b_lds(const __bf16* base, int kt, int lane) {
  return *(const v16bf*)(base + ((size_t)(kt * 32 + lane)) * 16);  // 32 B/lane
}

// Pack a [K x HID] fp32 weight's 16-column slice into LDS in B-fragment order.
// B layout (16-bit, 32x16): lanes 0-15: N=lane,    K = kt*32 + (e<8 ? e : 8+e)
//                           lanes16-31: N=lane-16, K = kt*32 + 8 + (e<8 ? e : 8+e)
__device__ __forceinline__ void pack_weights(__bf16* dst, const float* W,
                                             int col0, int ktiles) {
  for (int idx = threadIdx.x; idx < ktiles * 32; idx += TPB) {
    const int kt = idx >> 5, l = idx & 31;
    const int kbase = kt * 32 + ((l < 16) ? 0 : 8);
    const int col   = col0 + (l & 15);
    __bf16* o = dst + (size_t)idx * 16;
#pragma unroll
    for (int e = 0; e < 16; ++e)
      o[e] = (__bf16)W[(size_t)(kbase + e + ((e >= 8) ? 8 : 0)) * HID + col];
  }
}

__global__ void rnn_init_kernel(unsigned char* ws) {
  const int i = blockIdx.x * blockDim.x + threadIdx.x;
  if (i == 0) *(unsigned*)(ws + WS_CNT) = 0u;
  unsigned* h1 = (unsigned*)(ws + WS_H1B0);   // zero h1(-1), h2(-1)
  unsigned* h2 = (unsigned*)(ws + WS_H2B0);
  if (i < 131072 / 4) { h1[i] = 0u; h2[i] = 0u; }
}

__global__ void __launch_bounds__(TPB)
rnn_persistent_kernel(const int* __restrict__ x, const float* __restrict__ emb,
                      const float* __restrict__ Wi1, const float* __restrict__ Wh1,
                      const float* __restrict__ b1,
                      const float* __restrict__ Wi2, const float* __restrict__ Wh2,
                      const float* __restrict__ b2,
                      const float* __restrict__ Wd, const float* __restrict__ bd,
                      float* __restrict__ out, unsigned char* __restrict__ ws) {
  extern __shared__ __bf16 smem[];
  __bf16* sWi1 = smem + LDS_WI1;
  __bf16* sWh1 = smem + LDS_WH1;
  __bf16* sWi2 = smem + LDS_WI2;
  __bf16* sWh2 = smem + LDS_WH2;

  const int g     = blockIdx.x;           // column-slice id
  const int col0  = g * 16;
  const int lane  = threadIdx.x & 31;
  const int wv    = threadIdx.x >> 5;     // M-tile id
  const int row0  = wv * 16;
  const int myrow = row0 + (lane & 15);   // A-matrix M row this lane feeds
  const int mycol = col0 + (lane & 15);   // D-matrix N col this lane holds

  unsigned* cnt = (unsigned*)(ws + WS_CNT);
  __bf16* h1b[2] = { (__bf16*)(ws + WS_H1B0), (__bf16*)(ws + WS_H1B1) };
  __bf16* h2b[2] = { (__bf16*)(ws + WS_H2B0), (__bf16*)(ws + WS_H2B1) };

  // ---- prologue: pin this WG's weight slices in LDS (fragment layout) ----
  pack_weights(sWi1, Wi1, col0, 16);   // [512 x16]  bf16
  pack_weights(sWh1, Wh1, col0, 32);   // [1024x16]
  pack_weights(sWi2, Wi2, col0, 32);
  pack_weights(sWh2, Wh2, col0, 32);
  __syncthreads();

  const float bias1 = b1[mycol];
  const float bias2 = b2[mycol];

  unsigned nbar = 0;
  for (int t = 0; t < SEQ; ++t) {
    const __bf16* h1r = h1b[t & 1];          // h1(t-1)
    __bf16*       h1w = h1b[(t + 1) & 1];    // h1(t)
    const __bf16* h2r = h2b[t & 1];          // h2(t-1)
    __bf16*       h2w = h2b[(t + 1) & 1];    // h2(t)

    // ================= layer 1 =================
    const int tok = x[(size_t)myrow * SEQ + t];
    const float* erow = emb + (size_t)tok * EMBD;

    v8f acc = {};
#pragma unroll 4
    for (int kt = 0; kt < EMBD / 32; ++kt) {  // xe @ Wi1
      v16bf a = load_a_f32(erow, kt, lane);
      v16bf b = load_b_lds(sWi1, kt, lane);
      acc = __builtin_amdgcn_wmma_f32_16x16x32_bf16(false, a, false, b,
                                                    (short)0, acc, false, false);
    }
#pragma unroll 4
    for (int kt = 0; kt < HID / 32; ++kt) {   // h1 @ Wh1
      v16bf a = load_a_bf16(h1r, myrow, kt, lane);
      v16bf b = load_b_lds(sWh1, kt, lane);
      acc = __builtin_amdgcn_wmma_f32_16x16x32_bf16(false, a, false, b,
                                                    (short)0, acc, false, false);
    }
    // D layout: reg r -> M = (lane<16 ? r : 8+r), N = lane&15
#pragma unroll
    for (int r = 0; r < 8; ++r) {
      const int m = (lane < 16) ? r : (8 + r);
      h1w[(size_t)(row0 + m) * HID + mycol] = (__bf16)tanhf(acc[r] + bias1);
    }
    ++nbar; grid_sync(cnt, nbar * NWG);       // publish h1(t)

    // ================= layer 2 =================
    v8f acc2 = {};
#pragma unroll 4
    for (int kt = 0; kt < HID / 32; ++kt) {   // h1(t) @ Wi2
      v16bf a = load_a_bf16(h1w, myrow, kt, lane);
      v16bf b = load_b_lds(sWi2, kt, lane);
      acc2 = __builtin_amdgcn_wmma_f32_16x16x32_bf16(false, a, false, b,
                                                     (short)0, acc2, false, false);
    }
#pragma unroll 4
    for (int kt = 0; kt < HID / 32; ++kt) {   // h2(t-1) @ Wh2
      v16bf a = load_a_bf16(h2r, myrow, kt, lane);
      v16bf b = load_b_lds(sWh2, kt, lane);
      acc2 = __builtin_amdgcn_wmma_f32_16x16x32_bf16(false, a, false, b,
                                                     (short)0, acc2, false, false);
    }
#pragma unroll
    for (int r = 0; r < 8; ++r) {
      const int m = (lane < 16) ? r : (8 + r);
      h2w[(size_t)(row0 + m) * HID + mycol] = (__bf16)tanhf(acc2[r] + bias2);
    }
    ++nbar; grid_sync(cnt, nbar * NWG);       // publish h2(t)
  }

  // ---- head: logits = h2(S-1) @ Wd + bd ; sigmoid.  h2(511) lives in buf0.
  if (g == 0 && threadIdx.x < BATCH) {
    const __bf16* hf = h2b[SEQ & 1];          // SEQ=512 -> buf[(511+1)&1]=buf0
    float z = bd[0];
    for (int k = 0; k < HID; ++k)
      z += (float)hf[(size_t)threadIdx.x * HID + k] * Wd[k];
    out[threadIdx.x] = 1.0f / (1.0f + expf(-z));
  }
}

extern "C" void kernel_launch(void* const* d_in, const int* in_sizes, int n_in,
                              void* d_out, int out_size, void* d_ws, size_t ws_size,
                              hipStream_t stream) {
  (void)in_sizes; (void)n_in; (void)out_size; (void)ws_size;
  const int*   x   = (const int*)  d_in[0];
  const float* emb = (const float*)d_in[1];
  const float* Wi1 = (const float*)d_in[2];
  const float* Wh1 = (const float*)d_in[3];
  const float* b1  = (const float*)d_in[4];
  const float* Wi2 = (const float*)d_in[5];
  const float* Wh2 = (const float*)d_in[6];
  const float* b2  = (const float*)d_in[7];
  const float* Wd  = (const float*)d_in[8];
  const float* bd  = (const float*)d_in[9];

  unsigned char* ws = (unsigned char*)d_ws;
  rnn_init_kernel<<<128, 256, 0, stream>>>(ws);

  const size_t smem_bytes = (size_t)LDS_HALVES * sizeof(__bf16);  // 114688 B
  rnn_persistent_kernel<<<NWG, TPB, smem_bytes, stream>>>(
      x, emb, Wi1, Wh1, b1, Wi2, Wh2, b2, Wd, bd, (float*)d_out, ws);
}
